// GraphSAGE_84464826843467
// MI455X (gfx1250) — compile-verified, matrix operand order checked by reading, not run on hardware
//
#include <hip/hip_runtime.h>

#define N_NODES 50000
#define N_EDGES 800000
#define D_IN    128
#define D_HID   256
#define D_OUT   128

typedef __attribute__((ext_vector_type(16))) __bf16 v16bf;
typedef __attribute__((ext_vector_type(8)))  __bf16 v8bf;
typedef __attribute__((ext_vector_type(8)))  float  v8f;
typedef __attribute__((ext_vector_type(4)))  unsigned int u32x4;
typedef __attribute__((ext_vector_type(8)))  int i32x8;
typedef __attribute__((ext_vector_type(4)))  int i32x4;

#if defined(__has_builtin)
#if __has_builtin(__builtin_amdgcn_tensor_load_to_lds) && \
    __has_builtin(__builtin_amdgcn_s_wait_tensorcnt)
#define USE_TDM 1
#endif
#endif
#ifndef USE_TDM
#define USE_TDM 0
#endif

__device__ __forceinline__ v16bf cat16(v8bf lo, v8bf hi) {
    return __builtin_shufflevector(lo, hi, 0, 1, 2, 3, 4, 5, 6, 7,
                                   8, 9, 10, 11, 12, 13, 14, 15);
}

// ---------------------------------------------------------------- utilities
__global__ void zero_f32(float* __restrict__ p, long n) {
    long i = (long)blockIdx.x * blockDim.x + threadIdx.x;
    long stride = (long)gridDim.x * blockDim.x;
    for (; i < n; i += stride) p[i] = 0.0f;
}

__global__ void degree_kernel(const int* __restrict__ dst, float* __restrict__ deg, int nE) {
    int e = blockIdx.x * blockDim.x + threadIdx.x;
    if (e < nE) atomicAdd(&deg[dst[e]], 1.0f);
}

__global__ void deginv_kernel(const float* __restrict__ deg, float* __restrict__ deginv, int n) {
    int i = blockIdx.x * blockDim.x + threadIdx.x;
    if (i < n) deginv[i] = 1.0f / fmaxf(deg[i], 1.0f);
}

// msgsum[dst] += feat[src]; one thread = 4 consecutive features of one edge
__global__ void scatter_kernel(const int* __restrict__ src, const int* __restrict__ dst,
                               const float* __restrict__ feat, float* __restrict__ msgsum,
                               int nE, int D) {
    long t = (long)blockIdx.x * blockDim.x + threadIdx.x;
    int per_edge = D >> 2;
    long total = (long)nE * per_edge;
    if (t >= total) return;
    int e  = (int)(t / per_edge);
    int f4 = (int)(t % per_edge);
    int s = src[e], d = dst[e];
    float4 v = *(const float4*)(feat + (size_t)s * D + f4 * 4);
    float* o = msgsum + (size_t)d * D + f4 * 4;
    atomicAdd(o + 0, v.x);
    atomicAdd(o + 1, v.y);
    atomicAdd(o + 2, v.z);
    atomicAdd(o + 3, v.w);
}

// f32 -> bf16, 8 elements per thread (packed 16B store)
__global__ void cvt_bf16(const float* __restrict__ in, __bf16* __restrict__ out, long n8) {
    long i = (long)blockIdx.x * blockDim.x + threadIdx.x;
    if (i >= n8) return;
    long base = i * 8;
    const float4* p = (const float4*)(in + base);
    float4 a = p[0], b = p[1];
    v8bf o;
    o[0] = (__bf16)a.x; o[1] = (__bf16)a.y; o[2] = (__bf16)a.z; o[3] = (__bf16)a.w;
    o[4] = (__bf16)b.x; o[5] = (__bf16)b.y; o[6] = (__bf16)b.z; o[7] = (__bf16)b.w;
    *(v8bf*)(out + base) = o;
}

// mean = msgsum * deginv[row], f32 -> bf16
template<int D>
__global__ void cvt_mean(const float* __restrict__ ms, const float* __restrict__ deginv,
                         __bf16* __restrict__ out, long n8) {
    long i = (long)blockIdx.x * blockDim.x + threadIdx.x;
    if (i >= n8) return;
    long base = i * 8;
    float sc = deginv[base / D];
    const float4* p = (const float4*)(ms + base);
    float4 a = p[0], b = p[1];
    v8bf o;
    o[0] = (__bf16)(a.x * sc); o[1] = (__bf16)(a.y * sc);
    o[2] = (__bf16)(a.z * sc); o[3] = (__bf16)(a.w * sc);
    o[4] = (__bf16)(b.x * sc); o[5] = (__bf16)(b.y * sc);
    o[6] = (__bf16)(b.z * sc); o[7] = (__bf16)(b.w * sc);
    *(v8bf*)(out + base) = o;
}

#if USE_TDM
// Tensor DMA: load a 64-row x K-bf16 tile of a row-major [Dout,K] matrix into
// LDS, inserting 16B padding after every K*2-byte row (matches the bank-shift
// layout the ds_load fragment reads expect). CDNA5 ISA.
template<int K>
__device__ __forceinline__ void tdm_load_tile(const __bf16* gsrc, unsigned ldsByteOff) {
    unsigned long long ga = (unsigned long long)(size_t)gsrc;
    u32x4 g0;
    g0[0] = 1u;                                   // count=1, user mode
    g0[1] = ldsByteOff;                           // lds_addr (bytes)
    g0[2] = (unsigned)(ga & 0xffffffffu);         // global_addr[31:0]
    g0[3] = (unsigned)((ga >> 32) & 0x1ffffffu)   // global_addr[56:32]
            | 0x80000000u;                        // type=2 ("image")
    constexpr unsigned padInterval = (K == 128) ? 5u : 6u;  // 64 / 128 DWORD rows
    i32x8 g1;
    g1[0] = (int)((1u << 16)            // data_size = 2 bytes
                | (1u << 20)            // pad_enable
                | (padInterval << 22)   // pad every row
                | (3u << 25));          // pad_amount = 4 DWORDs (16B)
    g1[1] = (int)(((unsigned)K & 0xffffu) << 16);   // tensor_dim0[15:0]=K
    g1[2] = (int)(64u << 16);                       // tensor_dim1[15:0]=64
    g1[3] = (int)(((unsigned)K) << 16);             // tile_dim0=K
    g1[4] = 64;                                     // tile_dim1=64, tile_dim2=0
    g1[5] = K;                                      // tensor_dim0_stride[31:0]=K
    g1[6] = 0;
    g1[7] = 0;
    i32x4 gz = {0, 0, 0, 0};                        // groups 2/3 unused (2D tile)
#if __clang_major__ >= 23
    i32x8 gz8 = {};
    __builtin_amdgcn_tensor_load_to_lds(g0, g1, gz, gz, gz8, 0);
#else
    __builtin_amdgcn_tensor_load_to_lds(g0, g1, gz, gz, 0);
#endif
}
#endif

// --------------------------------------------------------------- WMMA GEMM
// out[M,Dout] = A1 @ Wl^T + bias + A2 @ Wr^T (optional ReLU). All A/W in bf16,
// A1 pre-scaled by deg_inv. Block = 256 thr = 8 waves; block owns a 64-col
// N-tile whose Wl/Wr rows are staged in LDS via the Tensor Data Mover (padded
// +16B/row: 4-bank shift per column -> conflict-free 16-lane b128 reads).
// Blocks persist over M strips so staging is amortized.
template<int K>
__global__ __launch_bounds__(256)
void sage_gemm(const __bf16* __restrict__ A1, const __bf16* __restrict__ A2,
               const __bf16* __restrict__ Wl, const __bf16* __restrict__ Wr,
               const float* __restrict__ bias, float* __restrict__ out,
               int M, int Dout, int do_relu) {
    constexpr int STRIDE = K + 8;          // bf16 elems per LDS column row
    extern __shared__ __bf16 smem[];       // [2][64][STRIDE], dyn-LDS offset 0
    __bf16* sWl = smem;
    __bf16* sWr = smem + 64 * STRIDE;

    const int lane = threadIdx.x & 31;
    const int wave = threadIdx.x >> 5;
    const int nBase = blockIdx.y * 64;

#if USE_TDM
    if (wave == 0) {
        tdm_load_tile<K>(Wl + (size_t)nBase * K, 0u);
        tdm_load_tile<K>(Wr + (size_t)nBase * K, (unsigned)(64 * STRIDE * 2));
        __builtin_amdgcn_s_wait_tensorcnt(0);
    }
#else
    {
        const int chunksPerCol = K / 8;
        const int totalChunks = 64 * chunksPerCol;
        for (int c = threadIdx.x; c < totalChunks; c += blockDim.x) {
            int col = c / chunksPerCol, k8 = c % chunksPerCol;
            *(v8bf*)(sWl + col * STRIDE + k8 * 8) =
                *(const v8bf*)(Wl + (size_t)(nBase + col) * K + k8 * 8);
            *(v8bf*)(sWr + col * STRIDE + k8 * 8) =
                *(const v8bf*)(Wr + (size_t)(nBase + col) * K + k8 * 8);
        }
    }
#endif
    __syncthreads();

    // per-lane fragment geometry (ISA 7.12.2, 16-bit 16x32 A layout)
    const int colInTile = lane & 15;
    const int kb = (lane >> 4) << 3;       // 0 or 8
    const __bf16* blp = sWl + colInTile * STRIDE + kb;
    const __bf16* brp = sWr + colInTile * STRIDE + kb;

    float biasv[4];
#pragma unroll
    for (int nt = 0; nt < 4; nt++) biasv[nt] = bias[nBase + nt * 16 + colInTile];

    const int strips = (M + 127) >> 7;
    for (int s = blockIdx.x; s < strips; s += gridDim.x) {
        const int mBase = s * 128 + wave * 16;
        int rowA = mBase + colInTile;
        int rA = rowA < M ? rowA : M - 1;   // clamp keeps EXEC all-ones for WMMA
        const __bf16* a1p = A1 + (size_t)rA * K + kb;
        const __bf16* a2p = A2 + (size_t)rA * K + kb;

        // prefetch next strip's A rows (gfx1250 global_prefetch_b8)
        int sn = s + gridDim.x;
        if (sn < strips) {
            int rn = mBase + colInTile + (sn - s) * 128;
            if (rn >= M) rn = M - 1;
            __builtin_prefetch(A1 + (size_t)rn * K, 0, 3);
            __builtin_prefetch(A2 + (size_t)rn * K, 0, 3);
        }

        v8f acc[4] = {};
#pragma unroll
        for (int k0 = 0; k0 < K; k0 += 32) {
            v16bf a1 = cat16(*(const v8bf*)(a1p + k0), *(const v8bf*)(a1p + k0 + 16));
            v16bf a2 = cat16(*(const v8bf*)(a2p + k0), *(const v8bf*)(a2p + k0 + 16));
            v16bf b1[4], b2[4];
#pragma unroll
            for (int nt = 0; nt < 4; nt++) {
                const __bf16* p1 = blp + nt * 16 * STRIDE + k0;
                const __bf16* p2 = brp + nt * 16 * STRIDE + k0;
                b1[nt] = cat16(*(const v8bf*)(p1), *(const v8bf*)(p1 + 16));
                b2[nt] = cat16(*(const v8bf*)(p2), *(const v8bf*)(p2 + 16));
            }
            // 4 independent WMMAs per group: no acc-chain between neighbors
#pragma unroll
            for (int nt = 0; nt < 4; nt++)
                acc[nt] = __builtin_amdgcn_wmma_f32_16x16x32_bf16(
                    false, a1, false, b1[nt], (short)0, acc[nt], false, false);
#pragma unroll
            for (int nt = 0; nt < 4; nt++)
                acc[nt] = __builtin_amdgcn_wmma_f32_16x16x32_bf16(
                    false, a2, false, b2[nt], (short)0, acc[nt], false, false);
        }

        // C/D layout: lane<16 -> M=mBase+v, lane>=16 -> M=mBase+8+v; N = colInTile
        const int rbase = mBase + kb;
#pragma unroll
        for (int nt = 0; nt < 4; nt++) {
            int col = nBase + nt * 16 + colInTile;
#pragma unroll
            for (int v = 0; v < 8; v++) {
                int row = rbase + v;
                if (row < M) {
                    float val = acc[nt][v] + biasv[nt];
                    if (do_relu) val = fmaxf(val, 0.0f);
                    out[(size_t)row * Dout + col] = val;
                }
            }
        }
    }
}

// ---------------------------------------------------------------- dispatch
extern "C" void kernel_launch(void* const* d_in, const int* in_sizes, int n_in,
                              void* d_out, int out_size, void* d_ws, size_t ws_size,
                              hipStream_t stream) {
    const float* x    = (const float*)d_in[0];
    const float* Wl1  = (const float*)d_in[1];
    const float* bl1  = (const float*)d_in[2];
    const float* Wr1  = (const float*)d_in[3];
    const float* Wl2  = (const float*)d_in[4];
    const float* bl2  = (const float*)d_in[5];
    const float* Wr2  = (const float*)d_in[6];
    const int*   edge = (const int*)d_in[7];   // JAX default x64-off: int32
    const int* src = edge;
    const int* dst = edge + N_EDGES;
    float* out = (float*)d_out;

    char* w = (char*)d_ws;
    float* deg    = (float*)w;  w += (size_t)N_NODES * 4;
    float* deginv = (float*)w;  w += (size_t)N_NODES * 4;
    float* msgsum = (float*)w;  w += (size_t)N_NODES * D_HID * 4;   // reused
    float* h      = (float*)w;  w += (size_t)N_NODES * D_HID * 4;
    __bf16* xbf   = (__bf16*)w; w += (size_t)N_NODES * D_IN * 2;
    __bf16* hbf   = (__bf16*)w; w += (size_t)N_NODES * D_HID * 2;
    __bf16* mbf   = (__bf16*)w; w += (size_t)N_NODES * D_HID * 2;   // reused
    __bf16* Wl1b  = (__bf16*)w; w += (size_t)D_HID * D_IN * 2;
    __bf16* Wr1b  = (__bf16*)w; w += (size_t)D_HID * D_IN * 2;
    __bf16* Wl2b  = (__bf16*)w; w += (size_t)D_OUT * D_HID * 2;
    __bf16* Wr2b  = (__bf16*)w; w += (size_t)D_OUT * D_HID * 2;

    // weight conversions (tiny, independent)
    cvt_bf16<<<16, 256, 0, stream>>>(Wl1, Wl1b, (long)D_HID * D_IN / 8);
    cvt_bf16<<<16, 256, 0, stream>>>(Wr1, Wr1b, (long)D_HID * D_IN / 8);
    cvt_bf16<<<16, 256, 0, stream>>>(Wl2, Wl2b, (long)D_OUT * D_HID / 8);
    cvt_bf16<<<16, 256, 0, stream>>>(Wr2, Wr2b, (long)D_OUT * D_HID / 8);

    // degrees
    zero_f32<<<512, 256, 0, stream>>>(deg, N_NODES);
    zero_f32<<<4096, 256, 0, stream>>>(msgsum, (long)N_NODES * D_IN);
    degree_kernel<<<(N_EDGES + 255) / 256, 256, 0, stream>>>(dst, deg, N_EDGES);
    deginv_kernel<<<(N_NODES + 255) / 256, 256, 0, stream>>>(deg, deginv, N_NODES);

    // ---- layer 1 ----
    {
        long total = (long)N_EDGES * (D_IN / 4);
        scatter_kernel<<<(int)((total + 255) / 256), 256, 0, stream>>>(
            src, dst, x, msgsum, N_EDGES, D_IN);
        long n8 = (long)N_NODES * D_IN / 8;
        cvt_bf16<<<(int)((n8 + 255) / 256), 256, 0, stream>>>(x, xbf, n8);
        cvt_mean<D_IN><<<(int)((n8 + 255) / 256), 256, 0, stream>>>(msgsum, deginv, mbf, n8);
        dim3 g(128, D_HID / 64);
        size_t lds = 2ull * 64 * (D_IN + 8) * sizeof(__bf16);
        sage_gemm<D_IN><<<g, 256, lds, stream>>>(
            mbf, xbf, Wl1b, Wr1b, bl1, h, N_NODES, D_HID, 1);
    }

    // ---- layer 2 ----
    {
        zero_f32<<<8192, 256, 0, stream>>>(msgsum, (long)N_NODES * D_HID);
        long total = (long)N_EDGES * (D_HID / 4);
        scatter_kernel<<<(int)((total + 255) / 256), 256, 0, stream>>>(
            src, dst, h, msgsum, N_EDGES, D_HID);
        long n8 = (long)N_NODES * D_HID / 8;
        cvt_bf16<<<(int)((n8 + 255) / 256), 256, 0, stream>>>(h, hbf, n8);
        cvt_mean<D_HID><<<(int)((n8 + 255) / 256), 256, 0, stream>>>(msgsum, deginv, mbf, n8);
        dim3 g(128, D_OUT / 64);
        size_t lds = 2ull * 64 * (D_HID + 8) * sizeof(__bf16);
        sage_gemm<D_HID><<<g, 256, lds, stream>>>(
            mbf, hbf, Wl2b, Wr2b, bl2, out, N_NODES, D_OUT, 0);
    }
}
